// PatchInpainting_42528766165299
// MI455X (gfx1250) — compile-verified, matrix operand order checked by reading, not run on hardware
//
#include <hip/hip_runtime.h>
#include <hip/hip_bf16.h>
#include <stdint.h>

typedef __attribute__((ext_vector_type(16))) __bf16 v16bf;
typedef __attribute__((ext_vector_type(8)))  float  v8f;
typedef __attribute__((ext_vector_type(4)))  unsigned int v4u;
typedef __attribute__((ext_vector_type(8)))  int v8i;
typedef __attribute__((ext_vector_type(4)))  int v4i;

#define B_    2
#define LQ_   2048
#define LK_   2048
#define E_    1024
#define H_    16
#define DK_   64
#define DV_   64
#define TOPK_ 64
#define HD_   1024   // H*DK == H*DV

#define LOGITS_BYTES (16 * LK_ * 4)          // 128 KB logits tile
#define STAGE_TILE_BYTES 2048                // 16 rows x 64 bf16
#define STAGE_BYTES (8 * 2 * STAGE_TILE_BYTES)  // 8 waves x double buffer

// ---------- helpers ----------

static __device__ __forceinline__ unsigned short f2bf(float x) {
    unsigned int b = __float_as_uint(x);
    b += 0x7FFFu + ((b >> 16) & 1u);          // round-to-nearest-even
    return (unsigned short)(b >> 16);
}

// 16-bit A/B WMMA fragment for 16x16x32: each lane holds two contiguous
// 8-element K-chunks of its row/column: p[0..7] and p[16..23].
static __device__ __forceinline__ v16bf load_frag16(const unsigned short* p) {
    union { v16bf v; uint4 q[2]; } f;
    f.q[0] = *(const uint4*)(p);
    f.q[1] = *(const uint4*)(p + 16);
    return f.v;
}

static __device__ __forceinline__ v8f wmma_bf16(v16bf a, v16bf b, v8f c) {
    return __builtin_amdgcn_wmma_f32_16x16x32_bf16(
        /*neg_a=*/false, a, /*neg_b=*/false, b,
        /*c_mod=*/(short)0, c, /*reuse_a=*/false, /*reuse_b=*/false);
}

static __device__ __forceinline__ float wave_max(float v) {
#pragma unroll
    for (int m = 16; m >= 1; m >>= 1) v = fmaxf(v, __shfl_xor(v, m, 32));
    return v;
}
static __device__ __forceinline__ float wave_min(float v) {
#pragma unroll
    for (int m = 16; m >= 1; m >>= 1) v = fminf(v, __shfl_xor(v, m, 32));
    return v;
}
static __device__ __forceinline__ float wave_sum(float v) {
#pragma unroll
    for (int m = 16; m >= 1; m >>= 1) v += __shfl_xor(v, m, 32);
    return v;
}
static __device__ __forceinline__ int wave_sumi(int v) {
#pragma unroll
    for (int m = 16; m >= 1; m >>= 1) v += __shfl_xor(v, m, 32);
    return v;
}

// ---------- Tensor Data Mover: async 2-D bf16 tile load into LDS ----------
// Builds a D# per CDNA5 ISA (group0 + group1, groups 2/3
// disabled -> 2-D tensor). data_size=2B, tile [tile_d1 rows x tile_d0 cols],
// row stride = stride0 elements. Tracked with TENSORcnt.
static __device__ __forceinline__ void tdm_load_tile_bf16(
    unsigned lds_byte_addr, unsigned long long gaddr,
    unsigned tensor_d0, unsigned tensor_d1,
    unsigned tile_d0, unsigned tile_d1, unsigned stride0)
{
    v4u g0;
    g0.x = 1u;                                          // count=1, user load
    g0.y = lds_byte_addr;                               // lds_addr
    g0.z = (unsigned)(gaddr & 0xffffffffull);           // global_addr lo
    g0.w = (unsigned)((gaddr >> 32) & 0x01ffffffull)    // global_addr hi (57b)
         | (2u << 30);                                  // type = 2 ("image")
    v8i g1;
    g1[0] = (int)(1u << 16);                            // data_size = 2 bytes
    g1[1] = (int)((tensor_d0 & 0xffffu) << 16);         // tensor_dim0[15:0]
    g1[2] = (int)(((tensor_d0 >> 16) & 0xffffu) |
                  ((tensor_d1 & 0xffffu) << 16));       // dim0 hi | dim1 lo
    g1[3] = (int)(((tensor_d1 >> 16) & 0xffffu) |
                  ((tile_d0 & 0xffffu) << 16));         // dim1 hi | tile_dim0
    g1[4] = (int)(tile_d1 & 0xffffu);                   // tile_dim1, tile_dim2=0
    g1[5] = (int)stride0;                               // tensor_dim0_stride lo
    g1[6] = 0;
    g1[7] = 0;
    v4i z4 = {0, 0, 0, 0};
#if defined(__clang_major__) && (__clang_major__ >= 23)
    v8i z8 = {0, 0, 0, 0, 0, 0, 0, 0};
    __builtin_amdgcn_tensor_load_to_lds(g0, g1, z4, z4, z8, 0);
#else
    __builtin_amdgcn_tensor_load_to_lds(g0, g1, z4, z4, 0);
#endif
}

// ---------- elementwise f32 -> bf16 ----------

__global__ void f32_to_bf16_kernel(const float* __restrict__ src,
                                   unsigned short* __restrict__ dst, int n) {
    int i = blockIdx.x * blockDim.x + threadIdx.x;
    if (i < n) dst[i] = f2bf(src[i]);
}

// ---------- generic WMMA GEMM: C[M,N] = alpha * A[M,K] * W[N,K]^T ----------
// mode 0: f32 out row-major (Cf)
// mode 1: bf16 out row-major (Cb)
// mode 2: bf16 out scattered as vpT[B,H,DV,LK]  (M = B*LK, N = H*DV)

__global__ __launch_bounds__(256)
void wmma_gemm_bf16(const unsigned short* __restrict__ A,
                    const unsigned short* __restrict__ W,
                    float* __restrict__ Cf,
                    unsigned short* __restrict__ Cb,
                    int M, int N, int K, float alpha, int mode) {
    const int lane = threadIdx.x & 31;
    const int wave = threadIdx.x >> 5;
    const int hl   = lane >> 4;
    const int l16  = lane & 15;
    const int ntiles = N >> 4;
    const int tiles  = (M >> 4) * ntiles;
    int tile = blockIdx.x * 8 + wave;
    if (tile >= tiles) return;
    const int mt = tile / ntiles;
    const int nt = tile - mt * ntiles;

    const unsigned short* arow = A + (size_t)(mt * 16 + l16) * K + hl * 8;
    const unsigned short* brow = W + (size_t)(nt * 16 + l16) * K + hl * 8;

    v8f acc = {};
    for (int kb = 0; kb < K; kb += 32) {
        if (kb + 256 < K) {
            __builtin_prefetch(arow + kb + 256, 0, 3);
            __builtin_prefetch(brow + kb + 256, 0, 3);
        }
        v16bf a = load_frag16(arow + kb);
        v16bf b = load_frag16(brow + kb);
        acc = wmma_bf16(a, b, acc);
    }

#pragma unroll
    for (int r = 0; r < 8; ++r) {
        int m = mt * 16 + r + hl * 8;
        int n = nt * 16 + l16;
        float v = acc[r] * alpha;
        if (mode == 0) {
            Cf[(size_t)m * N + n] = v;
        } else if (mode == 1) {
            Cb[(size_t)m * N + n] = f2bf(v);
        } else {
            int b_ = m >> 11, kk = m & 2047;
            int h_ = n >> 6,  d  = n & 63;
            Cb[((size_t)((b_ * H_ + h_) * DV_ + d)) * LK_ + kk] = f2bf(v);
        }
    }
}

// ---------- fused attention: logits -> top-k -> softmax -> AV ----------
// One block = one (b, h, 16-query tile). LDS: 128 KB logits + 32 KB K staging.
// K tiles are staged into LDS by the Tensor Data Mover, double-buffered per
// wave, so WMMA on tile i overlaps the DMA of tile i+1.

__global__ __launch_bounds__(256)
void attn_topk_kernel(const unsigned short* __restrict__ qp,   // [B*LQ, H*DK] bf16 (pre-scaled)
                      const unsigned short* __restrict__ kp,   // [B*LK, H*DK] bf16
                      const unsigned short* __restrict__ vpT,  // [B, H, DV, LK] bf16
                      float* __restrict__ attn,                // [B, H, LQ, LK] f32
                      unsigned short* __restrict__ mixed) {    // [B*LQ, H*DV] bf16
    extern __shared__ char smraw[];
    float* sm = (float*)smraw;      // 16 x LK_ logits
    const int lane = threadIdx.x & 31;
    const int wave = threadIdx.x >> 5;
    const int hl   = lane >> 4;
    const int l16  = lane & 15;
    const int bid  = blockIdx.x;
    const int qt   = bid & 127;
    const int h    = (bid >> 7) & 15;
    const int b    = bid >> 11;

    // ---- phase 1: logits tile [16 x 2048] via TDM-staged WMMA ----
    v16bf aQ0, aQ1;
    {
        const unsigned short* qr =
            qp + (size_t)(b * LQ_ + qt * 16 + l16) * HD_ + h * DK_ + hl * 8;
        aQ0 = load_frag16(qr);        // K chunk 0..31
        aQ1 = load_frag16(qr + 32);   // K chunk 32..63
    }

    const unsigned stage_off0 = (unsigned)LOGITS_BYTES + wave * (2 * STAGE_TILE_BYTES);
    unsigned short* stage0 = (unsigned short*)(smraw + stage_off0);
    unsigned short* stage1 = (unsigned short*)(smraw + stage_off0 + STAGE_TILE_BYTES);
    const unsigned lds_base0 = (unsigned)(size_t)(void*)stage0;
    const unsigned lds_base1 = (unsigned)(size_t)(void*)stage1;
    const unsigned long long kp_base = (unsigned long long)(size_t)kp;

    // issue DMA for this wave's first K tile (kt = wave)
    {
        unsigned long long g = kp_base +
            2ull * ((unsigned long long)(b * LK_ + wave * 16) * HD_ + h * DK_);
        tdm_load_tile_bf16(lds_base0, g, HD_, B_ * LK_, DK_, 16, HD_);
    }

#pragma unroll 1
    for (int i = 0; i < 16; ++i) {
        const int kt = wave + i * 8;
        if (i + 1 < 16) {
            // prefetch next tile into the other buffer, wait for previous
            unsigned long long g = kp_base +
                2ull * ((unsigned long long)(b * LK_ + (kt + 8) * 16) * HD_ + h * DK_);
            tdm_load_tile_bf16((i & 1) ? lds_base0 : lds_base1, g,
                               HD_, B_ * LK_, DK_, 16, HD_);
            __builtin_amdgcn_s_wait_tensorcnt(1);
        } else {
            __builtin_amdgcn_s_wait_tensorcnt(0);
        }
        const unsigned short* st = (i & 1) ? stage1 : stage0;   // [16 x 64] bf16
        const unsigned short* kr = st + l16 * DK_ + hl * 8;
        v8f acc = {};
        acc = wmma_bf16(aQ0, load_frag16(kr), acc);
        acc = wmma_bf16(aQ1, load_frag16(kr + 32), acc);
#pragma unroll
        for (int r = 0; r < 8; ++r)
            sm[(size_t)(r + hl * 8) * LK_ + kt * 16 + l16] = acc[r];
    }
    __syncthreads();

    // ---- phase 2: per-row top-64 threshold (bisection) + softmax ----
    float* attn_base = attn + ((size_t)(b * H_ + h) * LQ_ + qt * 16) * LK_;
#pragma unroll
    for (int rr = 0; rr < 2; ++rr) {
        const int row = wave * 2 + rr;
        float* rowp = sm + (size_t)row * LK_;
        float mx = -3.4e38f, mn = 3.4e38f;
        for (int j = lane; j < LK_; j += 32) {
            float x = rowp[j];
            mx = fmaxf(mx, x);
            mn = fminf(mn, x);
        }
        mx = wave_max(mx);
        mn = wave_min(mn);
        // largest t with count(x >= t) >= TOPK
        float lo = mn, hi = mx;
        for (int it = 0; it < 32; ++it) {
            float mid = 0.5f * (lo + hi);
            int c = 0;
            for (int j = lane; j < LK_; j += 32) c += (rowp[j] >= mid) ? 1 : 0;
            c = wave_sumi(c);
            if (c >= TOPK_) lo = mid; else hi = mid;
        }
        const float thr = lo;
        float s = 0.f;
        for (int j = lane; j < LK_; j += 32) {
            float x = rowp[j];
            if (x >= thr) s += __expf(x - mx);
        }
        s = wave_sum(s);
        const float inv = 1.0f / s;
        float* orow = attn_base + (size_t)row * LK_;
        for (int j = lane; j < LK_; j += 32) {
            float x = rowp[j];
            float p = (x >= thr) ? __expf(x - mx) * inv : 0.0f;
            rowp[j] = p;          // reused in phase 3
            orow[j] = p;          // dense attn output
        }
    }
    __syncthreads();

    // ---- phase 3: mixed[16 x 64] = attnTile[16 x 2048] * vp[2048 x 64] ----
    if (wave < 4) {
        const int nt = wave;
        const unsigned short* vr =
            vpT + ((size_t)(b * H_ + h) * DV_ + nt * 16 + l16) * LK_ + hl * 8;
        v8f acc = {};
        for (int kb = 0; kb < LK_; kb += 32) {
            union { v16bf v; unsigned short u[16]; } af;
            const int cbase = kb + hl * 8;
            const float* ar = sm + (size_t)l16 * LK_ + cbase;
#pragma unroll
            for (int i = 0; i < 8; ++i) {
                af.u[i]     = f2bf(ar[i]);
                af.u[i + 8] = f2bf(ar[i + 16]);
            }
            v16bf bv = load_frag16(vr + kb);
            acc = wmma_bf16(af.v, bv, acc);
        }
#pragma unroll
        for (int r = 0; r < 8; ++r) {
            int q = qt * 16 + r + hl * 8;
            int n = nt * 16 + l16;
            mixed[(size_t)(b * LQ_ + q) * HD_ + h * DV_ + n] = f2bf(acc[r]);
        }
    }
}

// ---------- host-side launch ----------

extern "C" void kernel_launch(void* const* d_in, const int* in_sizes, int n_in,
                              void* d_out, int out_size, void* d_ws, size_t ws_size,
                              hipStream_t stream) {
    const float* q    = (const float*)d_in[0];   // [B, LQ, E]
    const float* k    = (const float*)d_in[1];   // [B, LK, E]
    const float* v    = (const float*)d_in[2];   // [B, LK, DV]
    const float* w_qs = (const float*)d_in[3];   // [H*DK, E]
    const float* w_ks = (const float*)d_in[4];   // [H*DK, E]
    const float* w_vs = (const float*)d_in[5];   // [H*DV, DV]
    const float* fc   = (const float*)d_in[6];   // [DV, H*DV]

    float* outp = (float*)d_out;                         // [B, LQ, DV]
    float* attn = outp + (size_t)B_ * LQ_ * DV_;         // [B, H, LQ, LK]

    uint8_t* ws = (uint8_t*)d_ws;
    size_t off = 0;
    auto carve = [&](size_t elems) {
        unsigned short* p = (unsigned short*)(ws + off);
        off += ((elems * sizeof(unsigned short) + 255) & ~(size_t)255);
        return p;
    };
    unsigned short* qBF   = carve((size_t)B_ * LQ_ * E_);
    unsigned short* kBF   = carve((size_t)B_ * LK_ * E_);
    unsigned short* vBF   = carve((size_t)B_ * LK_ * DV_);
    unsigned short* wqsBF = carve((size_t)HD_ * E_);
    unsigned short* wksBF = carve((size_t)HD_ * E_);
    unsigned short* wvsBF = carve((size_t)HD_ * DV_);
    unsigned short* fcBF  = carve((size_t)DV_ * HD_);
    unsigned short* qpBF  = carve((size_t)B_ * LQ_ * HD_);
    unsigned short* kpBF  = carve((size_t)B_ * LK_ * HD_);
    unsigned short* vpT   = carve((size_t)B_ * H_ * DV_ * LK_);
    unsigned short* mixBF = carve((size_t)B_ * LQ_ * HD_);

    auto cvt = [&](const float* s, unsigned short* d, size_t n) {
        f32_to_bf16_kernel<<<(unsigned)((n + 255) / 256), 256, 0, stream>>>(s, d, (int)n);
    };
    cvt(q,    qBF,   (size_t)B_ * LQ_ * E_);
    cvt(k,    kBF,   (size_t)B_ * LK_ * E_);
    cvt(v,    vBF,   (size_t)B_ * LK_ * DV_);
    cvt(w_qs, wqsBF, (size_t)HD_ * E_);
    cvt(w_ks, wksBF, (size_t)HD_ * E_);
    cvt(w_vs, wvsBF, (size_t)HD_ * DV_);
    cvt(fc,   fcBF,  (size_t)DV_ * HD_);

    const int M = B_ * LQ_;   // 4096
    // qp = (q @ w_qs.T) * 1/sqrt(DK), bf16 out
    {
        int tiles = (M / 16) * (HD_ / 16);   // 16384
        wmma_gemm_bf16<<<tiles / 8, 256, 0, stream>>>(
            qBF, wqsBF, nullptr, qpBF, M, HD_, E_, 0.125f, 1);
    }
    // kp = k @ w_ks.T, bf16 out
    {
        int tiles = (M / 16) * (HD_ / 16);
        wmma_gemm_bf16<<<tiles / 8, 256, 0, stream>>>(
            kBF, wksBF, nullptr, kpBF, M, HD_, E_, 1.0f, 1);
    }
    // vp = v @ w_vs.T, bf16 out transposed to [B,H,DV,LK]
    {
        int tiles = (M / 16) * (HD_ / 16);
        wmma_gemm_bf16<<<tiles / 8, 256, 0, stream>>>(
            vBF, wvsBF, nullptr, vpT, M, HD_, DV_, 1.0f, 2);
    }
    // fused attention: logits -> top-k softmax -> attn out + mixed
    {
        const size_t lds_bytes = (size_t)LOGITS_BYTES + STAGE_BYTES;  // 160 KB
        attn_topk_kernel<<<B_ * H_ * (LQ_ / 16), 256, lds_bytes, stream>>>(
            qpBF, kpBF, vpT, attn, mixBF);
    }
    // out = mixed @ fc.T, f32 out
    {
        int tiles = (M / 16) * (DV_ / 16);   // 1024
        wmma_gemm_bf16<<<tiles / 8, 256, 0, stream>>>(
            mixBF, fcBF, outp, nullptr, M, DV_, HD_, 1.0f, 0);
    }
}
